// APLinear_28467043237959
// MI455X (gfx1250) — compile-verified, matrix operand order checked by reading, not run on hardware
//
#include <hip/hip_runtime.h>
#include <hip/hip_bf16.h>
#include <stdint.h>

typedef __attribute__((ext_vector_type(16))) _Float16 v16h;
typedef __attribute__((ext_vector_type(8)))  _Float16 v8h;
typedef __attribute__((ext_vector_type(8)))  float    v8f;

// Problem dims (fixed by the reference)
#define M_DIM 8192   // B*S = 4*2048
#define N_DIM 4096   // OUT
#define K_DIM 4096   // IN
#define NGROUPS 32   // K / 128

// ---------------------------------------------------------------------------
// Kernel A: decode bit-plane packed qweight -> f16 dequantized W [N, K]
// One thread handles one (row n, 8-feature chunk): 8 plane bytes -> 8x8 bit
// transpose -> 8 f16 weights.
// ---------------------------------------------------------------------------
__global__ __launch_bounds__(256) void aplinear_dequant_kernel(
    const int* __restrict__ qweight,   // [8, N, K/32] int32
    const float* __restrict__ scale,   // [N, NGROUPS]
    const float* __restrict__ zero,    // [N, NGROUPS]
    _Float16* __restrict__ Wde)        // [N, K] f16
{
    const unsigned idx  = blockIdx.x * 256u + threadIdx.x;   // N * 512 tasks
    const unsigned n    = idx >> 9;
    const unsigned task = idx & 511u;
    const unsigned fb   = task << 3;                 // feature base (0..4088)

    // feature -> byte mapping (derived from make_new_byte_indices + byte order)
    const unsigned warp = fb >> 10;
    const unsigned c    = (fb >> 8) & 3u;
    const unsigned t    = (fb >> 3) & 31u;
    const unsigned word = warp * 32u + t;

    const unsigned char* qb = (const unsigned char*)qweight;
    const size_t planeStride = (size_t)N_DIM * (K_DIM / 32) * 4;       // bytes
    const size_t rowOff = ((size_t)n * (K_DIM / 32) + word) * 4 + (c ^ 3u);

    // Pack plane p's byte at u64 byte (7-p)
    uint64_t x = 0;
#pragma unroll
    for (int p = 0; p < 8; ++p) {
        uint64_t b = qb[(size_t)p * planeStride + rowOff];
        x |= b << (8 * (7 - p));
    }
    // 8x8 bit-matrix transpose (Hacker's Delight)
    uint64_t tt;
    tt = (x ^ (x >> 7))  & 0x00AA00AA00AA00AAull; x ^= tt ^ (tt << 7);
    tt = (x ^ (x >> 14)) & 0x0000CCCC0000CCCCull; x ^= tt ^ (tt << 14);
    tt = (x ^ (x >> 28)) & 0x00000000F0F0F0F0ull; x ^= tt ^ (tt << 28);
    // Now qw[b] = byte (7-b) of x.

    const unsigned g = fb >> 7;                       // 128-feature group
    const float sc = scale[n * NGROUPS + g];
    const float zp = 16.0f * zero[n * NGROUPS + g];   // zero * 2^(8-4)

    v8h w;
#pragma unroll
    for (int b = 0; b < 8; ++b) {
        unsigned q = (unsigned)(x >> (8 * (7 - b))) & 255u;
        w[b] = (_Float16)(sc * ((float)q - zp));
    }
    *(v8h*)(Wde + (size_t)n * K_DIM + fb) = w;
}

// ---------------------------------------------------------------------------
// Kernel B: Y[M,N] = X[M,K](f32) * Wde[N,K](f16)^T + bias, f16 WMMA, f32 acc
// 256 threads = 8 waves (2 over M x 4 over N). Block tile 128(M) x 256(N),
// wave tile 64x64 = 4x4 WMMA accumulators. K-step 32. LDS double-buffered.
// ---------------------------------------------------------------------------
__global__ __launch_bounds__(256) void aplinear_gemm_kernel(
    const float* __restrict__ x,        // [M, K]
    const _Float16* __restrict__ Wde,   // [N, K]
    const float* __restrict__ bias,     // [N]
    float* __restrict__ out)            // [M, N]
{
    __shared__ _Float16 Xs[2][128][40];   // 32 K-halves + 8 pad (80B stride)
    __shared__ _Float16 Ws[2][256][40];

    const int n0 = blockIdx.x * 256;
    const int m0 = blockIdx.y * 128;

    const int tid  = threadIdx.x;
    const int wave = tid >> 5;
    const int lane = tid & 31;
    const int wm   = wave & 1;   // 2 waves over M
    const int wn   = wave >> 1;  // 4 waves over N
    const int l15  = lane & 15;
    const int klo  = (lane >> 4) * 8;  // lanes 0-15: K{0-7,16-23}; 16-31: K{8-15,24-31}

    // X-tile loader: 8 threads x float4 cover 32 cols, 32 rows/pass, 4 passes
    const int xlr = tid >> 3;
    const int xlc = (tid & 7) * 4;
    // W-tile loader: 4 threads x v8h cover 32 halves, 64 rows/pass, 4 passes
    const int wlr = tid >> 2;
    const int wlc = (tid & 3) * 8;

    v8f acc[4][4] = {};

    // ---- stage first K tile into buffer 0 ----
    {
        const int k0 = 0;
#pragma unroll
        for (int pass = 0; pass < 4; ++pass) {
            const int row = xlr + pass * 32;
            const float4 v = *(const float4*)(x + (size_t)(m0 + row) * K_DIM + k0 + xlc);
            _Float16* dst = &Xs[0][row][xlc];
            dst[0] = (_Float16)v.x; dst[1] = (_Float16)v.y;
            dst[2] = (_Float16)v.z; dst[3] = (_Float16)v.w;
        }
#pragma unroll
        for (int pass = 0; pass < 4; ++pass) {
            const int row = wlr + pass * 64;
            v8h v = *(const v8h*)(Wde + (size_t)(n0 + row) * K_DIM + k0 + wlc);
            *(v8h*)&Ws[0][row][wlc] = v;
        }
    }

    const int NIT = K_DIM / 32;
    int buf = 0;
    for (int it = 0; it < NIT; ++it) {
        __syncthreads();   // staged buffer `buf` visible; prev compute done

        // ---- prefetch next K tile into the other buffer ----
        const int nb = buf ^ 1;
        if (it + 1 < NIT) {
            const int k0 = (it + 1) * 32;
#pragma unroll
            for (int pass = 0; pass < 4; ++pass) {
                const int row = xlr + pass * 32;
                const float4 v = *(const float4*)(x + (size_t)(m0 + row) * K_DIM + k0 + xlc);
                _Float16* dst = &Xs[nb][row][xlc];
                dst[0] = (_Float16)v.x; dst[1] = (_Float16)v.y;
                dst[2] = (_Float16)v.z; dst[3] = (_Float16)v.w;
            }
#pragma unroll
            for (int pass = 0; pass < 4; ++pass) {
                const int row = wlr + pass * 64;
                v8h v = *(const v8h*)(Wde + (size_t)(n0 + row) * K_DIM + k0 + wlc);
                *(v8h*)&Ws[nb][row][wlc] = v;
            }
        }

        // ---- fragments from current buffer ----
        v16h afrag[4], bfrag[4];
#pragma unroll
        for (int mt = 0; mt < 4; ++mt) {
            const _Float16* p = &Xs[buf][wm * 64 + mt * 16 + l15][0];
            union { v16h v; v8h h[2]; } u;
            u.h[0] = *(const v8h*)(p + klo);
            u.h[1] = *(const v8h*)(p + klo + 16);
            afrag[mt] = u.v;
        }
#pragma unroll
        for (int nt = 0; nt < 4; ++nt) {
            const _Float16* p = &Ws[buf][wn * 64 + nt * 16 + l15][0];
            union { v16h v; v8h h[2]; } u;
            u.h[0] = *(const v8h*)(p + klo);
            u.h[1] = *(const v8h*)(p + klo + 16);
            bfrag[nt] = u.v;
        }

        // ---- 16 WMMAs ----
#pragma unroll
        for (int mt = 0; mt < 4; ++mt)
#pragma unroll
            for (int nt = 0; nt < 4; ++nt)
                acc[mt][nt] = __builtin_amdgcn_wmma_f32_16x16x32_f16(
                    false, afrag[mt], false, bfrag[nt],
                    (short)0, acc[mt][nt], false, false);

        buf = nb;
    }

    // ---- epilogue: bias add + store ----
    const int mrow_hi = (lane >> 4) * 8;   // accumulator M offset per lane half
#pragma unroll
    for (int mt = 0; mt < 4; ++mt) {
#pragma unroll
        for (int nt = 0; nt < 4; ++nt) {
            const int n_glob = n0 + wn * 64 + nt * 16 + l15;
            const float bv = bias[n_glob];
#pragma unroll
            for (int r = 0; r < 8; ++r) {
                const int m_glob = m0 + wm * 64 + mt * 16 + r + mrow_hi;
                out[(size_t)m_glob * N_DIM + n_glob] = acc[mt][nt][r] + bv;
            }
        }
    }
}

extern "C" void kernel_launch(void* const* d_in, const int* in_sizes, int n_in,
                              void* d_out, int out_size, void* d_ws, size_t ws_size,
                              hipStream_t stream) {
    (void)in_sizes; (void)n_in; (void)out_size; (void)ws_size;

    const float*    x       = (const float*)d_in[0];
    const int*      qweight = (const int*)d_in[1];
    const float*    scale   = (const float*)d_in[2];
    const float*    zero    = (const float*)d_in[3];
    const float*    bias    = (const float*)d_in[4];
    float*          out     = (float*)d_out;

    _Float16* Wde = (_Float16*)d_ws;   // N_DIM * K_DIM * 2 = 32 MB scratch

    // Pass 1: decode qweight -> f16 W (N*512 tasks)
    {
        const int total = N_DIM * (K_DIM / 8);
        aplinear_dequant_kernel<<<total / 256, 256, 0, stream>>>(qweight, scale, zero, Wde);
    }
    // Pass 2: WMMA GEMM + bias
    {
        dim3 grid(N_DIM / 256, M_DIM / 128);
        aplinear_gemm_kernel<<<grid, 256, 0, stream>>>(x, Wde, bias, out);
    }
}
